// FoundationalTimeSeriesModel_1391569404387
// MI455X (gfx1250) — matrix operand order repaired and checked by reading, not compiled, not run on hardware
//
#include <hip/hip_runtime.h>
#include <math.h>

// ---------------------------------------------------------------------------
// Problem constants (from reference setup_inputs)
// ---------------------------------------------------------------------------
constexpr int C_B  = 16;    // batch
constexpr int C_L  = 512;   // sequence length
constexpr int C_M  = 32;    // sensors
constexpr int C_BM = C_B * C_M;   // 512 sequences
constexpr int C_P  = 64;    // TCN channels (== T)
constexpr int C_D  = 128;   // transformer dim
constexpr int C_E  = 16;    // experts
constexpr int C_EH = 512;   // expert hidden
constexpr int C_O  = 128;   // expert out
constexpr int C_NH = 8;     // heads
constexpr int C_DH = 16;    // head dim
constexpr int BDIM = 256;   // 8 waves per block (wave32)

typedef __attribute__((ext_vector_type(16))) _Float16 v16h;
typedef __attribute__((ext_vector_type(8)))  _Float16 v8h;
typedef __attribute__((ext_vector_type(8)))  float    v8f;

// ---------------------------------------------------------------------------
// WMMA helpers (gfx1250: V_WMMA_F32_16X16X32_F16, wave32)
// ---------------------------------------------------------------------------
__device__ __forceinline__ v8f wmma_step(v16h a, v16h b, v8f c) {
  return __builtin_amdgcn_wmma_f32_16x16x32_f16(false, a, false, b, (short)0, c, false, false);
}

// A fragment: 16x32 f16, row-major source with leading dim ld.
// lane t: row m (passed in as absolute row), cols [c0+(t>>4)*8 .. +7] and +16..
__device__ __forceinline__ v16h load_a_frag(const _Float16* __restrict__ base,
                                            int ld, int row, int c0) {
  int lane = threadIdx.x & 31;
  const _Float16* p = base + (long long)row * ld + c0 + ((lane >> 4) << 3);
  union { v16h v; v8h h[2]; } u;
  u.h[0] = *(const v8h*)(p);
  u.h[1] = *(const v8h*)(p + 16);
  return u.v;
}

// B fragment: pre-packed tile of 512 halfs, lane-major (16 contiguous per lane)
__device__ __forceinline__ v16h load_b_frag(const _Float16* tile) {
  int lane = threadIdx.x & 31;
  const _Float16* p = tile + lane * 16;
  union { v16h v; v8h h[2]; } u;
  u.h[0] = *(const v8h*)(p);
  u.h[1] = *(const v8h*)(p + 8);
  return u.v;
}

// ---------------------------------------------------------------------------
// Weight packing: f32 (K x N logical, arbitrary strides) -> f16 fragment tiles
// tile (kt,nt): dst[tile*512 + lane*16 + j] = B[kt*32 + (j>>3)*16 + (lane>>4)*8 + (j&7), nt*16 + (lane&15)]
// ---------------------------------------------------------------------------
__global__ void pack_kernel(const float* __restrict__ src, _Float16* __restrict__ dst,
                            int K, int N, int sK, int sN,
                            int batch, long long srcBS, long long dstBS) {
  int KT = (K + 31) / 32, NT = (N + 15) / 16;
  long long total = (long long)batch * KT * NT * 512;
  for (long long i = (long long)blockIdx.x * blockDim.x + threadIdx.x; i < total;
       i += (long long)gridDim.x * blockDim.x) {
    int pos  = (int)(i & 511);
    long long ti = i >> 9;
    int tile = (int)(ti % (KT * NT));
    int b    = (int)(ti / (KT * NT));
    int kt = tile / NT, nt = tile % NT;
    int lane = pos >> 4, j = pos & 15;
    int n  = lane & 15;
    int kk  = kt * 32 + ((j >> 3) << 4) + ((lane >> 4) << 3) + (j & 7);
    int col = nt * 16 + n;
    float val = (kk < K && col < N)
                    ? src[b * srcBS + (long long)kk * sK + (long long)col * sN]
                    : 0.f;
    dst[b * dstBS + (long long)tile * 512 + pos] = (_Float16)val;
  }
}

// ---------------------------------------------------------------------------
// Stage 1: h0[s,l,p] = x[b,l,m]*mask*w[p] + b[p] + PE(l,p)   (f16, layout s,l,P)
// ---------------------------------------------------------------------------
__global__ void encode_init_kernel(const float* __restrict__ x, const float* __restrict__ mask,
                                   const float* __restrict__ ew, const float* __restrict__ eb,
                                   _Float16* __restrict__ h0) {
  long long total = (long long)C_BM * C_L * C_P;
  for (long long i = (long long)blockIdx.x * blockDim.x + threadIdx.x; i < total;
       i += (long long)gridDim.x * blockDim.x) {
    int p = (int)(i & 63);
    long long r = i >> 6;
    int l = (int)(r & 511);
    int s = (int)(r >> 9);
    int b = s >> 5, m = s & 31;
    float xv = x[((long long)b * C_L + l) * C_M + m] * mask[b * C_M + m];
    float val = xv * ew[p] + eb[p];
    int i2 = p & ~1;
    float freq = __expf(-9.210340371976184f * (float)i2 / (float)C_P);
    float ang = (float)l * freq;
    val += (p & 1) ? __cosf(ang) : __sinf(ang);
    h0[i] = (_Float16)val;
  }
}

__global__ void last_val_kernel(const float* __restrict__ x, const float* __restrict__ mask,
                                float* __restrict__ lastv) {
  int s = blockIdx.x * blockDim.x + threadIdx.x;
  if (s >= C_BM) return;
  int b = s >> 5, m = s & 31;
  lastv[s] = x[((long long)b * C_L + (C_L - 1)) * C_M + m] * mask[b * C_M + m];
}

// ---------------------------------------------------------------------------
// Stage 2: causal dilated conv as implicit GEMM with WMMA.
// Y[s,l,o] = relu( bias[o] + sum_{tap,cin} W[o,cin,tap] * X[s, l-(2-tap)*dil, cin] )
// RES variant: Y = relu(relu(conv) + res)
// Packed weights (24 tiles = 24KB) staged in LDS, shared by all 8 waves.
// one wave per (s, 16-row l-tile); per ktap: batch-load 2 A + 8 B then 8 WMMAs
// ---------------------------------------------------------------------------
template <bool RES>
__global__ void tcn_conv_kernel(const _Float16* __restrict__ X, const _Float16* __restrict__ Wpk,
                                const float* __restrict__ bias, const _Float16* __restrict__ res,
                                _Float16* __restrict__ Y, int dil) {
  __shared__ _Float16 wlds[24 * 512];
  {
    const uint4* src = (const uint4*)Wpk;
    uint4* dst = (uint4*)wlds;
    for (int i = threadIdx.x; i < (24 * 512) / 8; i += blockDim.x) dst[i] = src[i];
  }
  __syncthreads();

  int wid  = (blockIdx.x * blockDim.x + threadIdx.x) >> 5;
  int lane = threadIdx.x & 31;
  int nW   = (gridDim.x * blockDim.x) >> 5;
  int total = C_BM * (C_L / 16);
  v16h zf = {};
  for (int strip = wid; strip < total; strip += nW) {
    int s  = strip >> 5;
    int l0 = (strip & 31) << 4;
    const _Float16* Xs = X + (long long)s * C_L * C_P;
    v8f acc[4] = {};
#pragma unroll 1
    for (int ktap = 0; ktap < 3; ++ktap) {
      int shift = (2 - ktap) * dil;
      int row = l0 + (lane & 15) - shift;
      bool valid = (row >= 0);
      int rowc = valid ? row : 0;
      // batch-load both A fragments and all 8 B fragments, then run WMMA chain
      v16h a0 = load_a_frag(Xs, C_P, rowc, 0);
      v16h a1 = load_a_frag(Xs, C_P, rowc, 32);
      if (!valid) { a0 = zf; a1 = zf; }
      v16h bf[8];
#pragma unroll
      for (int i = 0; i < 8; ++i) bf[i] = load_b_frag(wlds + (ktap * 8 + i) * 512);
#pragma unroll
      for (int nt = 0; nt < 4; ++nt) acc[nt] = wmma_step(a0, bf[nt], acc[nt]);
#pragma unroll
      for (int nt = 0; nt < 4; ++nt) acc[nt] = wmma_step(a1, bf[4 + nt], acc[nt]);
    }
    int mb = (lane >> 4) * 8, n = lane & 15;
#pragma unroll
    for (int nt = 0; nt < 4; ++nt) {
      int col = nt * 16 + n;
      float bv = bias[col];
#pragma unroll
      for (int v = 0; v < 8; ++v) {
        int row = l0 + mb + v;
        long long oi = (long long)s * C_L * C_P + (long long)row * C_P + col;
        float d = fmaxf(acc[nt][v] + bv, 0.f);
        if (RES) d = fmaxf(d + (float)res[oi], 0.f);
        Y[oi] = (_Float16)d;
      }
    }
  }
}

// ---------------------------------------------------------------------------
// Stage 3: per-row LN over 64 channels, then mean over L -> pooled (f32 + f16)
// one wave per sequence; lane owns 2 channels
// ---------------------------------------------------------------------------
__global__ void ln_pool_kernel(const _Float16* __restrict__ h, const float* __restrict__ g,
                               const float* __restrict__ bta, const float* __restrict__ mask,
                               float* __restrict__ pooled32, _Float16* __restrict__ pooled16) {
  int s = blockIdx.x;
  int lane = threadIdx.x;
  const _Float16* hs = h + (long long)s * C_L * C_P;
  int c0 = lane * 2;
  float g0 = g[c0], g1 = g[c0 + 1], b0 = bta[c0], b1 = bta[c0 + 1];
  float acc0 = 0.f, acc1 = 0.f;
  for (int l = 0; l < C_L; ++l) {
    float x0 = (float)hs[l * C_P + c0];
    float x1 = (float)hs[l * C_P + c0 + 1];
    float sum = x0 + x1, sq = x0 * x0 + x1 * x1;
#pragma unroll
    for (int off = 16; off >= 1; off >>= 1) {
      sum += __shfl_xor(sum, off);
      sq  += __shfl_xor(sq, off);
    }
    float mean = sum * (1.f / 64.f);
    float var  = sq * (1.f / 64.f) - mean * mean;
    float rstd = rsqrtf(var + 1e-5f);
    acc0 += (x0 - mean) * rstd * g0 + b0;
    acc1 += (x1 - mean) * rstd * g1 + b1;
  }
  float mk = mask[s];
  float p0 = mk * acc0 * (1.f / (float)C_L);
  float p1 = mk * acc1 * (1.f / (float)C_L);
  pooled32[s * C_P + c0]     = p0;
  pooled32[s * C_P + c0 + 1] = p1;
  pooled16[s * C_P + c0]     = (_Float16)p0;
  pooled16[s * C_P + c0 + 1] = (_Float16)p1;
}

// ---------------------------------------------------------------------------
// Generic batched GEMM: out(M,N) = A(M,K) * Bpacked + bias, per-batch strides.
// flags: 1=RELU, 2=ADD into out32, 4=STORE out32, 8=STORE out16
// one wave per 16x64 strip (4 WMMA accumulators)
// ---------------------------------------------------------------------------
__global__ void gemm_kernel(const _Float16* __restrict__ A, const _Float16* __restrict__ Bpk,
                            const float* __restrict__ bias, int biasStride,
                            float* __restrict__ out32, _Float16* __restrict__ out16,
                            int Mrows, int N, int K, int flags, int batch,
                            long long aS, long long bS, long long oS) {
  int wid  = (blockIdx.x * blockDim.x + threadIdx.x) >> 5;
  int lane = threadIdx.x & 31;
  int nW   = (gridDim.x * blockDim.x) >> 5;
  int ng   = N >> 6;
  int spb  = (Mrows >> 4) * ng;
  long long total = (long long)batch * spb;
  for (long long sIdx = wid; sIdx < total; sIdx += nW) {
    int e  = (int)(sIdx / spb);
    int r  = (int)(sIdx % spb);
    int m0 = (r / ng) << 4;
    int n0 = (r % ng) << 6;
    const _Float16* Ab = A + (long long)e * aS;
    const _Float16* Bb = Bpk + (long long)e * bS;
    v8f acc[4] = {};
    int ktiles = K >> 5;
    int ntile  = N >> 4;
    int arow   = m0 + (lane & 15);
#pragma unroll 1
    for (int kt = 0; kt < ktiles; ++kt) {
      v16h a = load_a_frag(Ab, K, arow, kt << 5);
      v16h bf[4];
#pragma unroll
      for (int j = 0; j < 4; ++j)
        bf[j] = load_b_frag(Bb + ((long long)(kt * ntile + (n0 >> 4) + j) << 9));
#pragma unroll
      for (int j = 0; j < 4; ++j) acc[j] = wmma_step(a, bf[j], acc[j]);
    }
    int mb = (lane >> 4) * 8, nn = lane & 15;
#pragma unroll
    for (int j = 0; j < 4; ++j) {
      int col = n0 + j * 16 + nn;
      float bv = bias ? bias[(long long)e * biasStride + col] : 0.f;
#pragma unroll
      for (int v = 0; v < 8; ++v) {
        int row = m0 + mb + v;
        float d = acc[j][v] + bv;
        if (flags & 1) d = fmaxf(d, 0.f);
        long long oi = (long long)e * oS + (long long)row * N + col;
        if (flags & 2) out32[oi] += d;
        if (flags & 4) out32[oi] = d;
        if (flags & 8) out16[oi] = (_Float16)d;
      }
    }
  }
}

// ---------------------------------------------------------------------------
// LN over D=128 (one wave per row, 4 channels/lane), optional mask, f32/f16 out
// ---------------------------------------------------------------------------
__global__ void ln_rows_kernel(const float* __restrict__ t, const float* __restrict__ g,
                               const float* __restrict__ bta, const float* __restrict__ mask,
                               float* __restrict__ out32, _Float16* __restrict__ out16, int rows) {
  int wid  = (blockIdx.x * blockDim.x + threadIdx.x) >> 5;
  int lane = threadIdx.x & 31;
  int nW   = (gridDim.x * blockDim.x) >> 5;
  for (int r = wid; r < rows; r += nW) {
    const float* tr = t + (long long)r * C_D;
    float xv[4], sum = 0.f, sq = 0.f;
#pragma unroll
    for (int i = 0; i < 4; ++i) {
      xv[i] = tr[lane + 32 * i];
      sum += xv[i];
      sq  += xv[i] * xv[i];
    }
#pragma unroll
    for (int off = 16; off >= 1; off >>= 1) {
      sum += __shfl_xor(sum, off);
      sq  += __shfl_xor(sq, off);
    }
    float mean = sum * (1.f / 128.f);
    float var  = sq * (1.f / 128.f) - mean * mean;
    float rstd = rsqrtf(var + 1e-5f);
    float mk = mask ? mask[r] : 1.f;
#pragma unroll
    for (int i = 0; i < 4; ++i) {
      int c = lane + 32 * i;
      float d = ((xv[i] - mean) * rstd * g[c] + bta[c]) * mk;
      if (out32) out32[(long long)r * C_D + c] = d;
      if (out16) out16[(long long)r * C_D + c] = (_Float16)d;
    }
  }
}

__global__ void add_pos_kernel(float* __restrict__ t, const float* __restrict__ pos) {
  int i = blockIdx.x * blockDim.x + threadIdx.x;
  if (i >= C_BM * C_D) return;
  int r = i >> 7, c = i & 127;
  t[i] += pos[(r & 31) * C_D + c];
}

// ---------------------------------------------------------------------------
// Attention: per (b, head) block of 32 lanes (lane = query token)
// ---------------------------------------------------------------------------
__global__ void attention_kernel(const _Float16* __restrict__ q, const _Float16* __restrict__ k,
                                 const _Float16* __restrict__ v, const float* __restrict__ mask,
                                 _Float16* __restrict__ attn) {
  int b = blockIdx.x >> 3;
  int h = blockIdx.x & 7;
  int m = threadIdx.x;  // 0..31
  int row = b * 32 + m;
  float qv[C_DH];
#pragma unroll
  for (int d = 0; d < C_DH; ++d) qv[d] = (float)q[(long long)row * C_D + h * C_DH + d];
  float s[32];
  float mx = -1e30f;
  for (int kk = 0; kk < 32; ++kk) {
    float dot = 0.f;
#pragma unroll
    for (int d = 0; d < C_DH; ++d)
      dot += qv[d] * (float)k[(long long)(b * 32 + kk) * C_D + h * C_DH + d];
    dot = dot * 0.25f + ((mask[b * 32 + kk] == 0.f) ? -1e9f : 0.f);
    s[kk] = dot;
    mx = fmaxf(mx, dot);
  }
  float denom = 0.f;
  for (int kk = 0; kk < 32; ++kk) { s[kk] = __expf(s[kk] - mx); denom += s[kk]; }
  float inv = 1.f / denom;
#pragma unroll
  for (int d = 0; d < C_DH; ++d) {
    float o = 0.f;
    for (int kk = 0; kk < 32; ++kk)
      o += s[kk] * (float)v[(long long)(b * 32 + kk) * C_D + h * C_DH + d];
    attn[(long long)row * C_D + h * C_DH + d] = (_Float16)(o * inv);
  }
}

// ---------------------------------------------------------------------------
// ctx reductions: mean (masked count) and masked max over M -> g = [mean|max]
// ---------------------------------------------------------------------------
__global__ void reduce_ctx_kernel(const float* __restrict__ ctx, const float* __restrict__ mask,
                                  float* __restrict__ meanc, float* __restrict__ g32,
                                  _Float16* __restrict__ g16) {
  int b = blockIdx.x;
  int c = threadIdx.x;  // 0..127
  float cnt = 0.f, sum = 0.f, mxv = -1e9f;
  for (int m = 0; m < C_M; ++m) {
    float mk = mask[b * C_M + m];
    float val = ctx[(long long)(b * C_M + m) * C_D + c];
    sum += val;
    cnt += mk;
    mxv = fmaxf(mxv, (mk > 0.f) ? val : -1e9f);
  }
  cnt = fmaxf(cnt, 1.f);
  float mean = sum / cnt;
  meanc[b * C_D + c] = mean;
  g32[b * 2 * C_D + c]        = mean;
  g32[b * 2 * C_D + C_D + c]  = mxv;
  g16[b * 2 * C_D + c]        = (_Float16)mean;
  g16[b * 2 * C_D + C_D + c]  = (_Float16)mxv;
}

// ---------------------------------------------------------------------------
// MoE gating: logits = gin @ gw + gb ; top-2 + softmax(vals)
// ---------------------------------------------------------------------------
__global__ void gate_topk_kernel(const float* __restrict__ gin, int ldin,
                                 const float* __restrict__ gw, const float* __restrict__ gb,
                                 int rows, int Kg, int* __restrict__ idxout,
                                 float* __restrict__ wout) {
  int r = blockIdx.x * blockDim.x + threadIdx.x;
  if (r >= rows) return;
  float logits[C_E];
  for (int e = 0; e < C_E; ++e) {
    float a = gb[e];
    for (int kk = 0; kk < Kg; ++kk) a += gin[(long long)r * ldin + kk] * gw[kk * C_E + e];
    logits[e] = a;
  }
  int i0 = 0;
  for (int e = 1; e < C_E; ++e) if (logits[e] > logits[i0]) i0 = e;
  int i1 = (i0 == 0) ? 1 : 0;
  for (int e = 0; e < C_E; ++e)
    if (e != i0 && logits[e] > logits[i1]) i1 = e;
  float e1 = __expf(logits[i1] - logits[i0]);
  float w0 = 1.f / (1.f + e1);
  idxout[r * 2] = i0; idxout[r * 2 + 1] = i1;
  wout[r * 2] = w0;   wout[r * 2 + 1] = e1 * w0;
}

__global__ void moe_select_kernel(const float* __restrict__ allo, const int* __restrict__ idx,
                                  const float* __restrict__ w, float* __restrict__ y, int rows) {
  int i = blockIdx.x * blockDim.x + threadIdx.x;
  if (i >= rows * C_O) return;
  int r = i >> 7, o = i & 127;
  int i0 = idx[r * 2], i1 = idx[r * 2 + 1];
  y[i] = w[r * 2]     * allo[((long long)i0 * rows + r) * C_O + o] +
         w[r * 2 + 1] * allo[((long long)i1 * rows + r) * C_O + o];
}

__global__ void build_expin_kernel(const _Float16* __restrict__ ctx16,
                                   const float* __restrict__ meanc,
                                   _Float16* __restrict__ expin) {
  int i = blockIdx.x * blockDim.x + threadIdx.x;
  if (i >= C_BM * 2 * C_D) return;
  int r = i >> 8, c = i & 255;
  expin[i] = (c < C_D) ? ctx16[(long long)r * C_D + c]
                       : (_Float16)meanc[(r >> 5) * C_D + (c - C_D)];
}

// ---------------------------------------------------------------------------
// Heads: pred (B,M,3), rca (B,M); output layout per b: [pred 96 | fail 3 | rca 32]
// ---------------------------------------------------------------------------
__global__ void heads_kernel(const float* __restrict__ pooled, const float* __restrict__ ctx,
                             const float* __restrict__ yf, const float* __restrict__ yrca,
                             const float* __restrict__ lastv, const float* __restrict__ predw,
                             const float* __restrict__ predb, const float* __restrict__ rcaw,
                             const float* __restrict__ rcab, float* __restrict__ out) {
  int r = blockIdx.x * blockDim.x + threadIdx.x;
  if (r >= C_BM) return;
  int b = r >> 5, m = r & 31;
  float p0 = predb[0], p1 = predb[1], p2 = predb[2];
  float rc = rcab[0];
  for (int i = 0; i < C_P; ++i) {
    float v = pooled[(long long)r * C_P + i];
    p0 += v * predw[i * 3 + 0]; p1 += v * predw[i * 3 + 1]; p2 += v * predw[i * 3 + 2];
    rc += v * rcaw[i];
  }
  for (int i = 0; i < C_D; ++i) {
    float v = ctx[(long long)r * C_D + i];
    int j = C_P + i;
    p0 += v * predw[j * 3 + 0]; p1 += v * predw[j * 3 + 1]; p2 += v * predw[j * 3 + 2];
    rc += v * rcaw[j];
  }
  for (int i = 0; i < C_O; ++i) {
    float vf = yf[b * C_O + i];
    float vr = yrca[(long long)r * C_O + i];
    int j = C_P + C_D + i;
    p0 += vf * predw[j * 3 + 0]; p1 += vf * predw[j * 3 + 1]; p2 += vf * predw[j * 3 + 2];
    rc += vr * rcaw[j];
  }
  float lv = lastv[r];
  out[b * 131 + m * 3 + 0] = p0 + lv;
  out[b * 131 + m * 3 + 1] = p1 + lv;
  out[b * 131 + m * 3 + 2] = p2 + lv;
  out[b * 131 + 99 + m] = rc;
}

__global__ void fail_head_kernel(const float* __restrict__ yfail, const float* __restrict__ fw,
                                 const float* __restrict__ fb, float* __restrict__ out) {
  int t = blockIdx.x * blockDim.x + threadIdx.x;
  if (t >= C_B * 3) return;
  int b = t / 3, j = t % 3;
  float a = fb[j];
  for (int i = 0; i < C_O; ++i) a += yfail[b * C_O + i] * fw[i * 3 + j];
  out[b * 131 + 96 + j] = a;
}

// ---------------------------------------------------------------------------
// Workspace layout
// ---------------------------------------------------------------------------
constexpr size_t SZ_H      = (size_t)C_BM * C_L * C_P * 2;
constexpr size_t OFF_H0    = 0;
constexpr size_t OFF_H1    = OFF_H0 + SZ_H;
constexpr size_t OFF_H2    = OFF_H1 + SZ_H;
constexpr size_t OFF_WCONV = OFF_H2 + SZ_H;                     // 8 convs * 24 tiles
constexpr size_t OFF_WPOOL = OFF_WCONV + (size_t)8 * 24 * 512 * 2;
constexpr size_t OFF_WLAYR = OFF_WPOOL + (size_t)16 * 512 * 2;  // 2 layers * 256 tiles
constexpr size_t OFF_WEXP1 = OFF_WLAYR + (size_t)2 * 256 * 512 * 2;
constexpr size_t OFF_WEXP2 = OFF_WEXP1 + (size_t)16 * 256 * 512 * 2;
constexpr size_t OFF_LASTV = OFF_WEXP2 + (size_t)16 * 128 * 512 * 2;
constexpr size_t OFF_PL32  = OFF_LASTV + (size_t)C_BM * 4;
constexpr size_t OFF_PL16  = OFF_PL32 + (size_t)C_BM * C_P * 4;
constexpr size_t OFF_T32   = OFF_PL16 + (size_t)C_BM * C_P * 2;
constexpr size_t OFF_Y16   = OFF_T32 + (size_t)C_BM * C_D * 4;
constexpr size_t OFF_Q16   = OFF_Y16 + (size_t)C_BM * C_D * 2;
constexpr size_t OFF_K16   = OFF_Q16 + (size_t)C_BM * C_D * 2;
constexpr size_t OFF_V16   = OFF_K16 + (size_t)C_BM * C_D * 2;
constexpr size_t OFF_A16   = OFF_V16 + (size_t)C_BM * C_D * 2;
constexpr size_t OFF_FF16  = OFF_A16 + (size_t)C_BM * C_D * 2;
constexpr size_t OFF_CTX32 = OFF_FF16 + (size_t)C_BM * 2 * C_D * 2;
constexpr size_t OFF_CTX16 = OFF_CTX32 + (size_t)C_BM * C_D * 4;
constexpr size_t OFF_MEANC = OFF_CTX16 + (size_t)C_BM * C_D * 2;
constexpr size_t OFF_G32   = OFF_MEANC + (size_t)C_B * C_D * 4;
constexpr size_t OFF_G16   = OFF_G32 + (size_t)C_B * 2 * C_D * 4;
constexpr size_t OFF_EXPIN = OFF_G16 + (size_t)C_B * 2 * C_D * 2;
constexpr size_t OFF_HH16  = OFF_EXPIN + (size_t)C_BM * 2 * C_D * 2;
constexpr size_t OFF_ALLOG = OFF_HH16 + (size_t)C_E * C_BM * C_EH * 2;
constexpr size_t OFF_ALLOR = OFF_ALLOG + (size_t)C_E * C_B * C_O * 4;
constexpr size_t OFF_IDXF  = OFF_ALLOR + (size_t)C_E * C_BM * C_O * 4;
constexpr size_t OFF_WF    = OFF_IDXF + (size_t)C_B * 2 * 4;
constexpr size_t OFF_IDXFL = OFF_WF + (size_t)C_B * 2 * 4;
constexpr size_t OFF_WFL   = OFF_IDXFL + (size_t)C_B * 2 * 4;
constexpr size_t OFF_IDXR  = OFF_WFL + (size_t)C_B * 2 * 4;
constexpr size_t OFF_WR    = OFF_IDXR + (size_t)C_BM * 2 * 4;
constexpr size_t OFF_YF    = OFF_WR + (size_t)C_BM * 2 * 4;
constexpr size_t OFF_YFAIL = OFF_YF + (size_t)C_B * C_O * 4;
constexpr size_t OFF_YRCA  = OFF_YFAIL + (size_t)C_B * C_O * 4;

// ---------------------------------------------------------------------------
// Host-side launch helpers
// ---------------------------------------------------------------------------
static void launch_pack(const float* src, _Float16* dst, int K, int N, int sK, int sN,
                        int batch, long long srcBS, long long dstBS, hipStream_t s) {
  int KT = (K + 31) / 32, NT = (N + 15) / 16;
  long long total = (long long)batch * KT * NT * 512;
  int blocks = (int)((total + BDIM - 1) / BDIM);
  if (blocks > 8192) blocks = 8192;
  pack_kernel<<<blocks, BDIM, 0, s>>>(src, dst, K, N, sK, sN, batch, srcBS, dstBS);
}

static void launch_gemm(const _Float16* A, const _Float16* Bp, const float* bias, int biasStride,
                        float* out32, _Float16* out16, int M, int N, int K, int flags, int batch,
                        long long aS, long long bS, long long oS, hipStream_t s) {
  long long strips = (long long)batch * (M / 16) * (N / 64);
  int blocks = (int)((strips + 7) / 8);
  if (blocks > 4096) blocks = 4096;
  gemm_kernel<<<blocks, BDIM, 0, s>>>(A, Bp, bias, biasStride, out32, out16, M, N, K, flags,
                                      batch, aS, bS, oS);
}

extern "C" void kernel_launch(void* const* d_in, const int* in_sizes, int n_in,
                              void* d_out, int out_size, void* d_ws, size_t ws_size,
                              hipStream_t stream) {
  const float* x           = (const float*)d_in[0];
  const float* sensor_mask = (const float*)d_in[1];
  const float* enc_proj_w  = (const float*)d_in[2];
  const float* enc_proj_b  = (const float*)d_in[3];
  const float* tcn_w1      = (const float*)d_in[4];
  const float* tcn_b1      = (const float*)d_in[5];
  const float* tcn_w2      = (const float*)d_in[6];
  const float* tcn_b2      = (const float*)d_in[7];
  const float* enc_norm_g  = (const float*)d_in[8];
  const float* enc_norm_b  = (const float*)d_in[9];
  const float* pool_proj_w = (const float*)d_in[10];
  const float* pool_proj_b = (const float*)d_in[11];
  const float* pos_inter   = (const float*)d_in[12];
  const float* t_wq = (const float*)d_in[13]; const float* t_bq = (const float*)d_in[14];
  const float* t_wk = (const float*)d_in[15]; const float* t_bk = (const float*)d_in[16];
  const float* t_wv = (const float*)d_in[17]; const float* t_bv = (const float*)d_in[18];
  const float* t_wo = (const float*)d_in[19]; const float* t_bo = (const float*)d_in[20];
  const float* t_ln1_g = (const float*)d_in[21]; const float* t_ln1_b = (const float*)d_in[22];
  const float* t_ln2_g = (const float*)d_in[23]; const float* t_ln2_b = (const float*)d_in[24];
  const float* t_ff1_w = (const float*)d_in[25]; const float* t_ff1_b = (const float*)d_in[26];
  const float* t_ff2_w = (const float*)d_in[27]; const float* t_ff2_b = (const float*)d_in[28];
  const float* t_out_g = (const float*)d_in[29]; const float* t_out_b = (const float*)d_in[30];
  const float* exp_w1 = (const float*)d_in[31]; const float* exp_b1 = (const float*)d_in[32];
  const float* exp_w2 = (const float*)d_in[33]; const float* exp_b2 = (const float*)d_in[34];
  const float* gate_f_w = (const float*)d_in[35]; const float* gate_f_b = (const float*)d_in[36];
  const float* gate_fail_w = (const float*)d_in[37]; const float* gate_fail_b = (const float*)d_in[38];
  const float* gate_rca_w = (const float*)d_in[39]; const float* gate_rca_b = (const float*)d_in[40];
  const float* pred_w = (const float*)d_in[41]; const float* pred_b = (const float*)d_in[42];
  const float* fail_w = (const float*)d_in[43]; const float* fail_b = (const float*)d_in[44];
  const float* rca_w = (const float*)d_in[45]; const float* rca_b = (const float*)d_in[46];

  char* ws = (char*)d_ws;
  _Float16* H0 = (_Float16*)(ws + OFF_H0);
  _Float16* H1 = (_Float16*)(ws + OFF_H1);
  _Float16* H2 = (_Float16*)(ws + OFF_H2);
  _Float16* WCONV = (_Float16*)(ws + OFF_WCONV);
  _Float16* WPOOL = (_Float16*)(ws + OFF_WPOOL);
  _Float16* WLAYR = (_Float16*)(ws + OFF_WLAYR);
  _Float16* WEXP1 = (_Float16*)(ws + OFF_WEXP1);
  _Float16* WEXP2 = (_Float16*)(ws + OFF_WEXP2);
  float* LASTV = (float*)(ws + OFF_LASTV);
  float* PL32 = (float*)(ws + OFF_PL32);
  _Float16* PL16 = (_Float16*)(ws + OFF_PL16);
  float* T32 = (float*)(ws + OFF_T32);
  _Float16* Y16 = (_Float16*)(ws + OFF_Y16);
  _Float16* Q16 = (_Float16*)(ws + OFF_Q16);
  _Float16* K16 = (_Float16*)(ws + OFF_K16);
  _Float16* V16 = (_Float16*)(ws + OFF_V16);
  _Float16* A16 = (_Float16*)(ws + OFF_A16);
  _Float16* FF16 = (_Float16*)(ws + OFF_FF16);
  float* CTX32 = (float*)(ws + OFF_CTX32);
  _Float16* CTX16 = (_Float16*)(ws + OFF_CTX16);
  float* MEANC = (float*)(ws + OFF_MEANC);
  float* G32 = (float*)(ws + OFF_G32);
  _Float16* G16 = (_Float16*)(ws + OFF_G16);
  _Float16* EXPIN = (_Float16*)(ws + OFF_EXPIN);
  _Float16* HH16 = (_Float16*)(ws + OFF_HH16);
  float* ALLOG = (float*)(ws + OFF_ALLOG);
  float* ALLOR = (float*)(ws + OFF_ALLOR);
  int* IDXF = (int*)(ws + OFF_IDXF);   float* WFv = (float*)(ws + OFF_WF);
  int* IDXFL = (int*)(ws + OFF_IDXFL); float* WFL = (float*)(ws + OFF_WFL);
  int* IDXR = (int*)(ws + OFF_IDXR);   float* WRv = (float*)(ws + OFF_WR);
  float* YF = (float*)(ws + OFF_YF);
  float* YFAIL = (float*)(ws + OFF_YFAIL);
  float* YRCA = (float*)(ws + OFF_YRCA);
  float* out = (float*)d_out;

  // ---- weight packing (f32 -> f16 fragment tiles) ----
  for (int conv = 0; conv < 2; ++conv) {
    const float* wsrc = conv ? tcn_w2 : tcn_w1;
    for (int tap = 0; tap < 3; ++tap) {
      // per level: K=cin(stride 3), N=cout(stride 192); 8 tiles per (conv,tap)
      launch_pack(wsrc + tap, WCONV + (size_t)(conv * 24 + tap * 8) * 512,
                  C_P, C_P, 3, C_P * 3, /*batch=*/4,
                  (long long)C_P * C_P * 3, (long long)48 * 512, stream);
    }
  }
  launch_pack(pool_proj_w, WPOOL, C_P, C_D, C_D, 1, 1, 0, 0, stream);
  for (int l = 0; l < 2; ++l) {
    _Float16* lb = WLAYR + (size_t)l * 256 * 512;
    launch_pack(t_wq + (size_t)l * C_D * C_D, lb + 0 * 512, C_D, C_D, C_D, 1, 1, 0, 0, stream);
    launch_pack(t_wk + (size_t)l * C_D * C_D, lb + 32 * 512, C_D, C_D, C_D, 1, 1, 0, 0, stream);
    launch_pack(t_wv + (size_t)l * C_D * C_D, lb + 64 * 512, C_D, C_D, C_D, 1, 1, 0, 0, stream);
    launch_pack(t_wo + (size_t)l * C_D * C_D, lb + 96 * 512, C_D, C_D, C_D, 1, 1, 0, 0, stream);
    launch_pack(t_ff1_w + (size_t)l * C_D * 2 * C_D, lb + 128 * 512, C_D, 2 * C_D, 2 * C_D, 1, 1, 0, 0, stream);
    launch_pack(t_ff2_w + (size_t)l * 2 * C_D * C_D, lb + 192 * 512, 2 * C_D, C_D, C_D, 1, 1, 0, 0, stream);
  }
  launch_pack(exp_w1, WEXP1, 2 * C_D, C_EH, C_EH, 1, C_E,
              (long long)2 * C_D * C_EH, (long long)256 * 512, stream);
  launch_pack(exp_w2, WEXP2, C_EH, C_O, C_O, 1, C_E,
              (long long)C_EH * C_O, (long long)128 * 512, stream);

  // ---- encoder init ----
  encode_init_kernel<<<8192, BDIM, 0, stream>>>(x, sensor_mask, enc_proj_w, enc_proj_b, H0);
  last_val_kernel<<<2, BDIM, 0, stream>>>(x, sensor_mask, LASTV);

  // ---- TCN (4 levels, ping-pong H0/H2 with H1 temp) ----
  _Float16* cur = H0;
  for (int lev = 0; lev < 4; ++lev) {
    int dil = 1 << lev;
    _Float16* dst = (cur == H0) ? H2 : H0;
    tcn_conv_kernel<false><<<2048, BDIM, 0, stream>>>(
        cur, WCONV + (size_t)(lev * 2 + 0) * 24 * 512, tcn_b1 + lev * C_P, nullptr, H1, dil);
    tcn_conv_kernel<true><<<2048, BDIM, 0, stream>>>(
        H1, WCONV + (size_t)(lev * 2 + 1) * 24 * 512, tcn_b2 + lev * C_P, cur, dst, dil);
    cur = dst;
  }

  // ---- layernorm + masked mean pool ----
  ln_pool_kernel<<<C_BM, 32, 0, stream>>>(cur, enc_norm_g, enc_norm_b, sensor_mask, PL32, PL16);

  // ---- t = pooled @ pool_proj + b + pos ----
  launch_gemm(PL16, WPOOL, pool_proj_b, 0, T32, nullptr, C_BM, C_D, C_P, /*STORE32*/4,
              1, 0, 0, 0, stream);
  add_pos_kernel<<<(C_BM * C_D) / BDIM, BDIM, 0, stream>>>(T32, pos_inter);

  // ---- transformer layers ----
  for (int l = 0; l < 2; ++l) {
    const _Float16* lb = WLAYR + (size_t)l * 256 * 512;
    ln_rows_kernel<<<64, BDIM, 0, stream>>>(T32, t_ln1_g + l * C_D, t_ln1_b + l * C_D,
                                            nullptr, nullptr, Y16, C_BM);
    launch_gemm(Y16, lb + 0 * 512, t_bq + l * C_D, 0, nullptr, Q16, C_BM, C_D, C_D, 8, 1, 0, 0, 0, stream);
    launch_gemm(Y16, lb + 32 * 512, t_bk + l * C_D, 0, nullptr, K16, C_BM, C_D, C_D, 8, 1, 0, 0, 0, stream);
    launch_gemm(Y16, lb + 64 * 512, t_bv + l * C_D, 0, nullptr, V16, C_BM, C_D, C_D, 8, 1, 0, 0, 0, stream);
    attention_kernel<<<C_B * C_NH, 32, 0, stream>>>(Q16, K16, V16, sensor_mask, A16);
    launch_gemm(A16, lb + 96 * 512, t_bo + l * C_D, 0, T32, nullptr, C_BM, C_D, C_D, /*ADD*/2, 1, 0, 0, 0, stream);
    ln_rows_kernel<<<64, BDIM, 0, stream>>>(T32, t_ln2_g + l * C_D, t_ln2_b + l * C_D,
                                            nullptr, nullptr, Y16, C_BM);
    launch_gemm(Y16, lb + 128 * 512, t_ff1_b + l * 2 * C_D, 0, nullptr, FF16,
                C_BM, 2 * C_D, C_D, /*RELU|ST16*/9, 1, 0, 0, 0, stream);
    launch_gemm(FF16, lb + 192 * 512, t_ff2_b + l * C_D, 0, T32, nullptr,
                C_BM, C_D, 2 * C_D, /*ADD*/2, 1, 0, 0, 0, stream);
  }

  // ---- ctx = LN(t) * mask ; reductions ----
  ln_rows_kernel<<<64, BDIM, 0, stream>>>(T32, t_out_g, t_out_b, sensor_mask, CTX32, CTX16, C_BM);
  reduce_ctx_kernel<<<C_B, C_D, 0, stream>>>(CTX32, sensor_mask, MEANC, G32, G16);

  // ---- MoE over g (shared experts for y_f / y_fail) ----
  gate_topk_kernel<<<1, 32, 0, stream>>>(G32, 2 * C_D, gate_f_w, gate_f_b, C_B, 2 * C_D, IDXF, WFv);
  gate_topk_kernel<<<1, 32, 0, stream>>>(G32, 2 * C_D, gate_fail_w, gate_fail_b, C_B, 2 * C_D, IDXFL, WFL);
  launch_gemm(G16, WEXP1, exp_b1, C_EH, nullptr, HH16, C_B, C_EH, 2 * C_D, /*RELU|ST16*/9,
              C_E, 0, (long long)256 * 512, (long long)C_B * C_EH, stream);
  launch_gemm(HH16, WEXP2, exp_b2, C_O, ALLOG, nullptr, C_B, C_O, C_EH, /*ST32*/4,
              C_E, (long long)C_B * C_EH, (long long)128 * 512, (long long)C_B * C_O, stream);
  moe_select_kernel<<<(C_B * C_O + BDIM - 1) / BDIM, BDIM, 0, stream>>>(ALLOG, IDXF, WFv, YF, C_B);
  moe_select_kernel<<<(C_B * C_O + BDIM - 1) / BDIM, BDIM, 0, stream>>>(ALLOG, IDXFL, WFL, YFAIL, C_B);

  // ---- RCA MoE (512 tokens) ----
  build_expin_kernel<<<(C_BM * 2 * C_D) / BDIM, BDIM, 0, stream>>>(CTX16, MEANC, EXPIN);
  gate_topk_kernel<<<2, BDIM, 0, stream>>>(CTX32, C_D, gate_rca_w, gate_rca_b, C_BM, C_D, IDXR, WRv);
  launch_gemm(EXPIN, WEXP1, exp_b1, C_EH, nullptr, HH16, C_BM, C_EH, 2 * C_D, /*RELU|ST16*/9,
              C_E, 0, (long long)256 * 512, (long long)C_BM * C_EH, stream);
  launch_gemm(HH16, WEXP2, exp_b2, C_O, ALLOR, nullptr, C_BM, C_O, C_EH, /*ST32*/4,
              C_E, (long long)C_BM * C_EH, (long long)128 * 512, (long long)C_BM * C_O, stream);
  moe_select_kernel<<<(C_BM * C_O) / BDIM, BDIM, 0, stream>>>(ALLOR, IDXR, WRv, YRCA, C_BM);

  // ---- heads + output assembly ----
  heads_kernel<<<2, BDIM, 0, stream>>>(PL32, CTX32, YF, YRCA, LASTV, pred_w, pred_b, rca_w, rca_b, out);
  fail_head_kernel<<<1, 64, 0, stream>>>(YFAIL, fail_w, fail_b, out);

  (void)in_sizes; (void)n_in; (void)out_size; (void)ws_size;
}
